// semiCRF_61770219651515
// MI455X (gfx1250) — matrix operand order repaired and checked by reading, not compiled, not run on hardware
//
#include <hip/hip_runtime.h>
#include <hip/hip_bf16.h>
#include <stdint.h>

// ---------------------------------------------------------------------------
// Semi-CRF (B=64, T=1024, L=16, K=64) for gfx1250 (MI455X).
//
// partition kernel: 64 WGs (one per batch; >=64 WGPs stream the 256 MB feat
// tensor toward the 23.3 TB/s roofline) x 128 threads (4 waves).
//   Alpha rows cached as immutable f16 exp2(alpha_l - m_l) in an LDS ring
//   (m_l = running max at the row's birth). Per step:
//     G[l,k] = Aexp @ exp(trans)            [2x v_wmma_f32_16x16x32_f16/wave]
//     alpha[k] = mx + log2( sum_l exp2(feat*log2e + m_l - mx) * G[l,k] )
//   A-fragments are pure 4x ds_load_b128 (no per-step exp of old rows).
//   feat streamed 11 steps ahead via global_load_async_to_lds_b128 (ASYNCcnt).
//   17-slot rings + parity-buffered running max => 1 barrier per step.
//   raw v_exp_f32 / v_log_f32; cross-half join via v_permlanex16 (VALU only).
// ---------------------------------------------------------------------------

typedef __attribute__((ext_vector_type(16))) _Float16 v16h;
typedef __attribute__((ext_vector_type(8)))  float    v8f;
typedef __attribute__((ext_vector_type(4)))  int      v4i;

union ABfrag { v16h h16; v4i q[2]; };

#define TT      1024
#define LL      16
#define KK      64
#define NEG2    (-144269.5f)   // -100000 * log2(e)
#define LOG2E   1.4426950408889634f
#define LN2     0.6931471805599453f
#define RSLOTS  12             // feat LDS ring slots
#define LOOKA   11             // prefetch distance (steps)
#define ASLOTS  17             // ring slots (16 live + 1 write target)
#define ROWPAD  68             // f32 alpha row stride (bank spread)
#define AEPAD   72             // f16 Aexp row stride (144 B = 9*16, bank spread)

__device__ __forceinline__ float fexp2(float x) { return __builtin_amdgcn_exp2f(x); }
__device__ __forceinline__ float flog2(float x) { return __builtin_amdgcn_logf(x); }

// lane i <-> lane (i^16) swap, pure VALU (v_permlanex16_b32, identity selectors)
__device__ __forceinline__ float xhalf16(float x) {
  unsigned u = __float_as_uint(x);
  unsigned r = __builtin_amdgcn_permlanex16(u, u, 0x76543210u, 0xfedcba98u,
                                            false, false);
  return __uint_as_float(r);
}

__device__ __forceinline__ uint32_t lds_addr32(const void* p) {
  return (uint32_t)(uintptr_t)p;   // low 32 bits of generic addr = LDS offset
}

__device__ __forceinline__ void async_b128(uint32_t ldsOff, const float* gptr) {
  uint64_t ga = (uint64_t)(uintptr_t)gptr;
  asm volatile("global_load_async_to_lds_b128 %0, %1, off"
               :: "v"(ldsOff), "v"(ga)
               : "memory");
}

__device__ __forceinline__ void wait_asynccnt_20() {
  asm volatile("s_wait_asynccnt 20" ::: "memory");
}

__global__ void __launch_bounds__(128)
crf_partition_kernel(const float* __restrict__ feat,
                     const float* __restrict__ trans,
                     const unsigned char* __restrict__ mask,
                     float* __restrict__ out)
{
  __shared__ __align__(16) float    featRing[RSLOTS * LL * ROWPAD]; // 52224 B
  __shared__ __align__(16) float    alphaRing[ASLOTS * ROWPAD];     // log2 alpha
  __shared__ __align__(16) _Float16 aexpRing[ASLOTS * AEPAD];       // f16 exp rows
  __shared__ __align__(16) float    scaleRing[ASLOTS];              // birth max m_l
  __shared__ __align__(16) float    gw2[2][4];                      // running max
  __shared__ unsigned char mS[TT];

  const int b    = blockIdx.x;
  const int tid  = threadIdx.x;
  const int lane = tid & 31;
  const int w    = tid >> 5;   // wave id 0..3 -> k-tile
  const int n    = lane & 15;
  const int h    = lane >> 4;  // half: M 0..7 vs 8..15 in C, byte offset in A rows
  const float* featB = feat + (size_t)b * TT * LL * KK;

  // ---- constant B fragments: Texp = exp(trans), 32x16 f16 tiles, K split in 2
  v16h bfrag0, bfrag1;
#pragma unroll
  for (int e = 0; e < 16; ++e) {
    int j0 = ((lane >> 4) << 3) + (e & 7) + ((e >> 3) << 4); // K row within frag
    bfrag0[e] = (_Float16)__expf(trans[(j0     ) * KK + (w * 16 + n)]);
    bfrag1[e] = (_Float16)__expf(trans[(j0 + 32) * KK + (w * 16 + n)]);
  }

  // ---- phase 1 init: rings, mask, running max; prologue async prefetch
  for (int i = tid; i < ASLOTS * ROWPAD; i += 128) {
    int l = i / ROWPAD, k = i - l * ROWPAD;
    float v = NEG2;
    if (l == 0 && k < KK) v = featB[k] * LOG2E;
    alphaRing[i] = v;
  }
  for (int i = tid; i < ASLOTS * (AEPAD / 2); i += 128) ((int*)aexpRing)[i] = 0;
  for (int i = tid; i < ASLOTS; i += 128) scaleRing[i] = NEG2;
  for (int i = tid; i < TT; i += 128) mS[i] = mask[(size_t)b * TT + i];
  if (tid == 0) {
    float m0 = NEG2;
    for (int k = 0; k < KK; ++k) m0 = fmaxf(m0, featB[k] * LOG2E);
    scaleRing[0] = m0;
    for (int c = 0; c < 4; ++c) { gw2[0][c] = m0; gw2[1][c] = m0; }
  }
  const uint32_t ringBase = lds_addr32(&featRing[0]);
  for (int s = 1; s <= LOOKA; ++s) {
#pragma unroll
    for (int q = 0; q < 2; ++q) {
      int cc = tid * 2 + q;                 // 16-byte chunk id 0..255
      int l = cc >> 4, r = cc & 15;
      uint32_t lo = ringBase + (uint32_t)(((s % RSLOTS) * LL + l) * ROWPAD) * 4u
                             + (uint32_t)r * 16u;
      async_b128(lo, featB + (size_t)s * (LL * KK) + cc * 4);
    }
  }
  __syncthreads();

  // ---- phase 2 init: Aexp row 0 = exp2(alpha0 - m0)
  if (tid < KK) {
    aexpRing[tid] = (_Float16)fexp2(featB[tid] * LOG2E - scaleRing[0]);
  }
  __syncthreads();

  int hd0 = 0;                       // slot of row l=0 before current step
  for (int t = 1; t < TT; ++t) {
    // own step-t asyncs done (issued LOOKA iters ago / prologue)
    wait_asynccnt_20();
    __syncthreads();                 // all waves' step-t data + prev writes visible

    // issue prefetch for step t+LOOKA (after barrier: slot reuse is safe)
    {
      int s = t + LOOKA;
      int sA = s < TT ? s : (TT - 1);   // clamped addr; dead slots near the end
#pragma unroll
      for (int q = 0; q < 2; ++q) {
        int cc = tid * 2 + q;
        int l = cc >> 4, r = cc & 15;
        uint32_t lo = ringBase + (uint32_t)(((s % RSLOTS) * LL + l) * ROWPAD) * 4u
                               + (uint32_t)r * 16u;
        async_b128(lo, featB + (size_t)sA * (LL * KK) + cc * 4);
      }
    }

    const int par = t & 1;
    const float4 gv = *reinterpret_cast<const float4*>(&gw2[par][0]); // 1x b128
    const float mx = fmaxf(fmaxf(gv.x, gv.y), fmaxf(gv.z, gv.w));

    // ---- F-side: ef = exp2(feat*log2e + m_l - mx)  (overlaps WMMA)
    const float* fslot = &featRing[(t % RSLOTS) * LL * ROWPAD];
    const int k = w * 16 + n;
    float ef[8];
    float scl0 = 0.f;                       // on h==0 lanes: scale of row l=0
#pragma unroll
    for (int v = 0; v < 8; ++v) {
      int l = v + (h << 3);                 // C VGPR v holds M = v + 8h
      int sl = hd0 + l; if (sl >= ASLOTS) sl -= ASLOTS;
      float sc = scaleRing[sl];
      if (v == 0) scl0 = sc;
      ef[v] = fexp2(__builtin_fmaf(fslot[l * ROWPAD + k], LOG2E, sc - mx));
    }
    const float aOld = alphaRing[hd0 * ROWPAD + k];
    const bool  msk  = mS[t] != 0;

    // ---- A fragments: pure ds_load_b128 from immutable f16 rows
    int slotA = hd0 + n; if (slotA >= ASLOTS) slotA -= ASLOTS;
    const char* arow = (const char*)aexpRing + slotA * (AEPAD * 2) + h * 16;
    ABfrag fa0, fa1;
    fa0.q[0] = *(const v4i*)(arow +  0);    // j 0..7   (h=1: 8..15)
    fa0.q[1] = *(const v4i*)(arow + 32);    // j 16..23 (h=1: 24..31)
    fa1.q[0] = *(const v4i*)(arow + 64);    // j 32..39 (h=1: 40..47)
    fa1.q[1] = *(const v4i*)(arow + 96);    // j 48..55 (h=1: 56..63)

    // ---- G tile: 16x16 per wave, K=64 via two WMMAs
    v8f c = {};
    c = __builtin_amdgcn_wmma_f32_16x16x32_f16(false, fa0.h16, false, bfrag0,
                                               (short)0, c, false, false);
    c = __builtin_amdgcn_wmma_f32_16x16x32_f16(false, fa1.h16, false, bfrag1,
                                               (short)0, c, false, false);

    // ---- combine: p = sum_l ef[l] * G[l,k]  (FMA tree, 3 levels)
    float p01 = __builtin_fmaf(ef[0], c[0], ef[1] * c[1]);
    float p23 = __builtin_fmaf(ef[2], c[2], ef[3] * c[3]);
    float p45 = __builtin_fmaf(ef[4], c[4], ef[5] * c[5]);
    float p67 = __builtin_fmaf(ef[6], c[6], ef[7] * c[7]);
    float p   = (p01 + p23) + (p45 + p67);
    p += xhalf16(p);                        // join M halves (VALU permlanex16)

    const float aNew = mx + flog2(p);
    const float aOut = msk ? aNew : aOld;   // masked step re-pushes old row 0
    const float scSel = msk ? mx : scl0;    // birth scale of the pushed row

    // ---- shift-write: 17-slot rings => targets not read this step
    const int newHd = (hd0 == 0) ? (ASLOTS - 1) : (hd0 - 1);
    if (h == 0) {
      alphaRing[newHd * ROWPAD + k] = aOut;
      aexpRing[newHd * AEPAD + k]   = (_Float16)fexp2(aOut - scSel);
      // running max for next parity: fire-and-forget ds_max_num_f32
      atomicMax(&gw2[1 - par][w], aOut);
    }
    if (lane == 0) atomicMax(&gw2[1 - par][w], mx);
    if (tid == 0)  scaleRing[newHd] = scSel;
    hd0 = newHd;
  }

  __syncthreads();
  if (tid == 0) {
    float m = NEG2;
    for (int k = 0; k < KK; ++k) m = fmaxf(m, alphaRing[hd0 * ROWPAD + k]);
    float s = 0.f;
    for (int k = 0; k < KK; ++k) s += fexp2(alphaRing[hd0 * ROWPAD + k] - m);
    out[b] = out[b] - LN2 * (m + flog2(s));   // out[b] holds gold-path score
  }
}

// ---------------------------------------------------------------------------
__global__ void __launch_bounds__(64)
crf_score_kernel(const float* __restrict__ feat,
                 const float* __restrict__ trans,
                 const int* __restrict__ tags,
                 const unsigned char* __restrict__ mask,
                 float* __restrict__ out)
{
  __shared__ float trS[KK * KK];          // 16 KB
  __shared__ int   tagS[TT];
  __shared__ unsigned char mSS[TT];
  __shared__ unsigned char sidx[TT];
  __shared__ float partials[64];
  __shared__ float tsum;

  const int b = blockIdx.x, tid = threadIdx.x;
  for (int i = tid; i < KK * KK; i += 64) trS[i] = trans[i];
  for (int i = tid; i < TT; i += 64) {
    tagS[i] = tags[(size_t)b * TT + i];
    mSS[i]  = mask[(size_t)b * TT + i];
  }
  __syncthreads();

  if (tid == 0) {
    int lastTag = tagS[0], lastStep = 0;
    float ts = 0.f;
    for (int t = 1; t < TT; ++t) {
      int tg = tagS[t];
      int m  = mSS[t] ? 1 : 0;
      sidx[t] = (unsigned char)lastStep;
      if (m) ts += trS[lastTag * KK + tg];
      int newTag = m ? tg : lastTag;
      int ns = (lastTag == newTag) ? newTag : 0;
      if (ns > LL - 1) ns = LL - 1;
      lastTag = newTag; lastStep = ns;
    }
    tsum = ts;
  }
  __syncthreads();

  const float* featB = feat + (size_t)b * TT * LL * KK;
  float acc = 0.f;
  for (int t = 1 + tid; t < TT; t += 64) {
    if (mSS[t]) acc += featB[(size_t)t * (LL * KK) + (int)sidx[t] * KK + tagS[t]];
  }
  partials[tid] = acc;
  __syncthreads();
  if (tid == 0) {
    float s = featB[tagS[0]];             // feat[b,0,0,tag0]
    for (int i = 0; i < 64; ++i) s += partials[i];
    out[b] = s + tsum;
  }
}

// ---------------------------------------------------------------------------
extern "C" void kernel_launch(void* const* d_in, const int* in_sizes, int n_in,
                              void* d_out, int out_size, void* d_ws, size_t ws_size,
                              hipStream_t stream)
{
  const float*         feat  = (const float*)d_in[0];
  const float*         trans = (const float*)d_in[1];
  const int*           tags  = (const int*)d_in[2];
  const unsigned char* mask  = (const unsigned char*)d_in[3];
  float*               out   = (float*)d_out;

  crf_score_kernel<<<64, 64, 0, stream>>>(feat, trans, tags, mask, out);
  crf_partition_kernel<<<64, 128, 0, stream>>>(feat, trans, mask, out);
}
